// SetAbstractionStage_69054484185299
// MI455X (gfx1250) — matrix operand order repaired and checked by reading, not compiled, not run on hardware
//
#include <hip/hip_runtime.h>
#include <math.h>

// ---------------------------------------------------------------------------
// ParticleNet-style SetAbstraction for MI455X (gfx1250, wave32, WMMA).
// Phases: prep (bf16 weights) -> FPS -> KNN -> fused edge-build + conv1/conv2
// (v_wmma_f32_16x16x32_bf16) + BN/ReLU + attention softmax + weighted sums.
// ---------------------------------------------------------------------------

#define BATCH 16
#define NP    4096
#define NM    1024
#define NK    32
#define CIN   64
#define COUT  128
#define EDR   132      // real edge dim
#define ED    160      // padded to 5 * 32 for WMMA K-steps
#define FEPS  1e-8f
#define BNEPS 1e-5f
#define PI_F      3.14159265358979323846f
#define TWO_PI_F  6.28318530717958647692f

typedef __attribute__((ext_vector_type(16))) __bf16 v16bf;
typedef __attribute__((ext_vector_type(8)))  float  v8f;

union V16U { v16bf v; uint4 u[2]; };

__device__ __forceinline__ unsigned short f2bf(float f) {
  unsigned u = __float_as_uint(f);
  unsigned r = u + 0x7FFFu + ((u >> 16) & 1u);   // round-to-nearest-even
  return (unsigned short)(r >> 16);
}

// python-style mod(a-b+pi, 2pi) - pi  (result sign follows divisor)
__device__ __forceinline__ float dphi_f(float a, float b) {
  float x = a - b + PI_F;
  x -= floorf(x * (1.0f / TWO_PI_F)) * TWO_PI_F;
  return x - PI_F;
}

// ---------------------------------------------------------------------------
// Kernel 0: weight conversion to padded bf16 + folded BN scale/bias
// ---------------------------------------------------------------------------
__global__ void prep_kernel(const float* __restrict__ w1, const float* __restrict__ w2,
                            const float* __restrict__ g1, const float* __restrict__ be1,
                            const float* __restrict__ mu1, const float* __restrict__ va1,
                            const float* __restrict__ g2, const float* __restrict__ be2,
                            const float* __restrict__ mu2, const float* __restrict__ va2,
                            unsigned short* __restrict__ W1bf,
                            unsigned short* __restrict__ W2bf,
                            float* __restrict__ sb) {
  int t = threadIdx.x;
  for (int i = t; i < COUT * ED; i += blockDim.x) {
    int r = i / ED, c = i % ED;
    W1bf[i] = (c < EDR) ? f2bf(w1[r * EDR + c]) : (unsigned short)0;
  }
  for (int i = t; i < COUT * COUT; i += blockDim.x) W2bf[i] = f2bf(w2[i]);
  if (t < COUT) {
    float inv1 = g1[t] * rsqrtf(va1[t] + BNEPS);
    sb[t]          = inv1;
    sb[COUT + t]   = be1[t] - mu1[t] * inv1;
    float inv2 = g2[t] * rsqrtf(va2[t] + BNEPS);
    sb[2*COUT + t] = inv2;
    sb[3*COUT + t] = be2[t] - mu2[t] * inv2;
  }
}

// ---------------------------------------------------------------------------
// Kernel 1: farthest point sampling. One block per batch, 1024 threads,
// each thread owns 4 points; two-level (shfl + LDS) argmax, lowest-index ties.
// ---------------------------------------------------------------------------
__global__ __launch_bounds__(1024) void fps_kernel(const float* __restrict__ coords,
                                                   int* __restrict__ cidx) {
  __shared__ float rv[32];
  __shared__ int   ri[32];
  __shared__ int   s_cur;
  const int b = blockIdx.x, t = threadIdx.x;
  const float* eta = coords + (size_t)b * 2 * NP;
  const float* phi = eta + NP;
  float pe[4], pp[4], md[4];
#pragma unroll
  for (int j = 0; j < 4; ++j) {
    int p = t + j * 1024;
    pe[j] = eta[p]; pp[j] = phi[p]; md[j] = INFINITY;
  }
  int cur = 0;  // argmax of all-true mask == index 0
  for (int it = 0; it < NM; ++it) {
    if (t == 0) cidx[b * NM + it] = cur;
    float ce = eta[cur], cp = phi[cur];
    float bv = -INFINITY; int bi = 0x7fffffff;
#pragma unroll
    for (int j = 0; j < 4; ++j) {
      float de = pe[j] - ce, dp = dphi_f(pp[j], cp);
      float d = de * de + dp * dp;
      md[j] = fminf(md[j], d);
      if (md[j] > bv) { bv = md[j]; bi = t + j * 1024; }
    }
#pragma unroll
    for (int off = 16; off > 0; off >>= 1) {
      float ov = __shfl_down(bv, off, 32);
      int   oi = __shfl_down(bi, off, 32);
      if (ov > bv || (ov == bv && oi < bi)) { bv = ov; bi = oi; }
    }
    if ((t & 31) == 0) { rv[t >> 5] = bv; ri[t >> 5] = bi; }
    __syncthreads();
    if (t < 32) {
      float v2 = rv[t]; int i2 = ri[t];
#pragma unroll
      for (int off = 16; off > 0; off >>= 1) {
        float ov = __shfl_down(v2, off, 32);
        int   oi = __shfl_down(i2, off, 32);
        if (ov > v2 || (ov == v2 && oi < i2)) { v2 = ov; i2 = oi; }
      }
      if (t == 0) s_cur = i2;
    }
    __syncthreads();
    cur = s_cur;
  }
}

// ---------------------------------------------------------------------------
// Kernel 2: cross-set KNN (k=32 smallest). One block per (b, m).
// Distances in LDS, 32 rounds of masked argmin (lowest-index ties).
// Selection order is irrelevant downstream (softmax aggregation over k).
// ---------------------------------------------------------------------------
__global__ __launch_bounds__(128) void knn_kernel(const float* __restrict__ coords,
                                                  const int* __restrict__ cidx,
                                                  int* __restrict__ nidx) {
  __shared__ float sd[NP];
  __shared__ float rv[4];
  __shared__ int   ri[4];
  const int m = blockIdx.x, b = blockIdx.y, t = threadIdx.x;
  const float* eta = coords + (size_t)b * 2 * NP;
  const float* phi = eta + NP;
  const int cv = cidx[b * NM + m];
  const float ce = eta[cv], cp = phi[cv];
  for (int p = t; p < NP; p += 128) {
    float de = eta[p] - ce, dp = dphi_f(phi[p], cp);
    sd[p] = de * de + dp * dp;
  }
  __syncthreads();
  for (int r = 0; r < NK; ++r) {
    float bv = INFINITY; int bi = 0x7fffffff;
#pragma unroll 4
    for (int i = 0; i < NP / 128; ++i) {
      int p = t + i * 128;
      float v = sd[p];
      if (v < bv || (v == bv && p < bi)) { bv = v; bi = p; }
    }
#pragma unroll
    for (int off = 16; off > 0; off >>= 1) {
      float ov = __shfl_down(bv, off, 32);
      int   oi = __shfl_down(bi, off, 32);
      if (ov < bv || (ov == bv && oi < bi)) { bv = ov; bi = oi; }
    }
    if ((t & 31) == 0) { rv[t >> 5] = bv; ri[t >> 5] = bi; }
    __syncthreads();
    if (t == 0) {
      float v2 = rv[0]; int i2 = ri[0];
      for (int q = 1; q < 4; ++q)
        if (rv[q] < v2 || (rv[q] == v2 && ri[q] < i2)) { v2 = rv[q]; i2 = ri[q]; }
      nidx[((size_t)b * NM + m) * NK + r] = i2;
      sd[i2] = INFINITY;
    }
    __syncthreads();
  }
}

// ---------------------------------------------------------------------------
// Kernel 3: fused edge-build + WMMA conv1/conv2 + BN/ReLU + attention.
// 128 threads = 4 waves; wave w owns output channels [32w, 32w+32).
// Edge/h stored n-major in LDS so B-operand K-pairs are contiguous (b128 ds
// loads); W tiles stream from L2-resident bf16 weights (b128 global loads).
// ---------------------------------------------------------------------------
__global__ __launch_bounds__(128) void fused_kernel(
    const float* __restrict__ coords, const float* __restrict__ feats,
    const float* __restrict__ lvs, const int* __restrict__ cidx,
    const int* __restrict__ nidx,
    const unsigned short* __restrict__ W1bf, const unsigned short* __restrict__ W2bf,
    const float* __restrict__ sb, const float* __restrict__ watt,
    float* __restrict__ out) {
  __shared__ unsigned short s_edge[NK * ED];   // [n][c] bf16, padded to 160
  __shared__ unsigned short s_h[NK * COUT];    // [n][c] bf16
  __shared__ float s_msgs[NK * COUT];          // [n][c] f32
  __shared__ float s_cf[CIN];
  __shared__ float s_clv[4];
  __shared__ float s_nlv[4 * NK];              // [j][k]
  __shared__ float s_attn[NK];
  __shared__ int   s_nidx[NK];

  const int  t  = threadIdx.x;
  const int  w  = t >> 5;            // wave id (uniform)
  const int  ln = t & 15;
  const bool hi = (t & 31) >= 16;
  const float* s1  = sb;
  const float* bb1 = sb + COUT;
  const float* s2  = sb + 2 * COUT;
  const float* bb2 = sb + 3 * COUT;

  const size_t OFF_LV = (size_t)BATCH * COUT * NM;       // out_feats size
  const size_t OFF_CC = OFF_LV + (size_t)BATCH * 4 * NM; // + out_lvs
  const size_t OFF_MK = OFF_CC + (size_t)BATCH * 2 * NM; // + c_coords

  for (int cid = blockIdx.x; cid < BATCH * NM; cid += gridDim.x) {
    const int b = cid >> 10, m = cid & (NM - 1);
    const int cv = cidx[cid];
    const float* fb = feats + (size_t)b * CIN * NP;
    const float* lb = lvs   + (size_t)b * 4 * NP;

    // ---- stage 1: centroid gather + neighbor indices + small outputs ----
    if (t < CIN)               s_cf[t]            = fb[t * NP + cv];
    else if (t < CIN + 4)      s_clv[t - CIN]     = lb[(t - CIN) * NP + cv];
    else if (t < CIN + 4 + NK) s_nidx[t - CIN - 4] = nidx[(size_t)cid * NK + (t - CIN - 4)];
    if (t == 100) out[OFF_CC + (size_t)b * 2 * NM + m]      = coords[(size_t)b * 2 * NP + cv];
    if (t == 101) out[OFF_CC + (size_t)b * 2 * NM + NM + m] = coords[(size_t)b * 2 * NP + NP + cv];
    if (t == 102) out[OFF_MK + (size_t)b * NM + m] = 1.0f;
    __syncthreads();

    // ---- stage 2a: pairwise Lorentz features (threads 0..31, one per k) ----
    if (t < NK) {
      const int p = s_nidx[t];
      float px = lb[p], py = lb[NP + p], pz = lb[2 * NP + p], e = lb[3 * NP + p];
      s_nlv[t] = px; s_nlv[NK + t] = py; s_nlv[2 * NK + t] = pz; s_nlv[3 * NK + t] = e;
      float cpx = s_clv[0], cpy = s_clv[1], cpz = s_clv[2], cee = s_clv[3];
      float pti  = sqrtf(fmaxf(cpx * cpx + cpy * cpy, FEPS));
      float rapi = 0.5f * log1pf(2.0f * cpz / fmaxf(cee - cpz, 1e-20f));
      float phii = atan2f(cpy, cpx);
      float ptj  = sqrtf(fmaxf(px * px + py * py, FEPS));
      float rapj = 0.5f * log1pf(2.0f * pz / fmaxf(e - pz, 1e-20f));
      float phij = atan2f(py, px);
      float de_ = rapi - rapj, dp_ = dphi_f(phii, phij);
      float delta   = sqrtf(de_ * de_ + dp_ * dp_);
      float lndelta = logf(fmaxf(delta, FEPS));
      float ptmin   = fminf(pti, ptj);
      float lnkt    = logf(fmaxf(ptmin * delta, FEPS));
      float lnz     = logf(fmaxf(ptmin / fmaxf(pti + ptj, FEPS), FEPS));
      float sx = cpx + px, sy = cpy + py, sz = cpz + pz, se = cee + e;
      float m2   = se * se - sx * sx - sy * sy - sz * sz;
      float lnm2 = logf(fmaxf(m2, FEPS));
      unsigned short* er = s_edge + t * ED;
      er[128] = f2bf(lnkt); er[129] = f2bf(lnz);
      er[130] = f2bf(lndelta); er[131] = f2bf(lnm2);
      for (int c = EDR; c < ED; ++c) er[c] = 0;  // zero K-padding
    }
    // ---- stage 2b: feature channels (all 128 threads, one channel each) ----
    {
      const int c = t;
      const float cfv = s_cf[c & (CIN - 1)];
      const float* frow = fb + (size_t)((c >= CIN) ? (c - CIN) : 0) * NP;
      for (int n = 0; n < NK; ++n) {
        float val = (c < CIN) ? cfv : (frow[s_nidx[n]] - cfv);
        s_edge[n * ED + c] = f2bf(val);
      }
    }
    __syncthreads();

    // ---- conv1: D(128x32) = W1(128x160) x edge(160x32), 5 K-steps ----
    v8f a00 = {0,0,0,0,0,0,0,0}, a01 = a00, a10 = a00, a11 = a00;
    const int rowb = w * 32;
#pragma unroll
    for (int kt = 0; kt < 5; ++kt) {
      const int kb = kt * 32;
      V16U A0, A1, B0, B1;
      int off = (rowb + ln) * ED + kb + (hi ? 8 : 0);
      A0.u[0] = *(const uint4*)(W1bf + off);
      A0.u[1] = *(const uint4*)(W1bf + off + 16);
      off = (rowb + 16 + ln) * ED + kb + (hi ? 8 : 0);
      A1.u[0] = *(const uint4*)(W1bf + off);
      A1.u[1] = *(const uint4*)(W1bf + off + 16);
      int boff = ln * ED + kb + (hi ? 8 : 0);
      B0.u[0] = *(const uint4*)(s_edge + boff);
      B0.u[1] = *(const uint4*)(s_edge + boff + 16);
      boff = (16 + ln) * ED + kb + (hi ? 8 : 0);
      B1.u[0] = *(const uint4*)(s_edge + boff);
      B1.u[1] = *(const uint4*)(s_edge + boff + 16);
      a00 = __builtin_amdgcn_wmma_f32_16x16x32_bf16(false, A0.v, false, B0.v, (short)0, a00, false, false);
      a01 = __builtin_amdgcn_wmma_f32_16x16x32_bf16(false, A0.v, false, B1.v, (short)0, a01, false, false);
      a10 = __builtin_amdgcn_wmma_f32_16x16x32_bf16(false, A1.v, false, B0.v, (short)0, a10, false, false);
      a11 = __builtin_amdgcn_wmma_f32_16x16x32_bf16(false, A1.v, false, B1.v, (short)0, a11, false, false);
    }
    // BN1 + ReLU, pack bf16, channel-contiguous b128 stores into s_h [n][c]
    {
      auto bn1_store = [&](v8f acc, int rt, int nt) {
        const int cb = rowb + rt * 16 + (hi ? 8 : 0);
        const int n  = nt * 16 + ln;
        unsigned pk[4];
#pragma unroll
        for (int q = 0; q < 4; ++q) {
          int c0 = cb + 2 * q, c1 = c0 + 1;
          float x0 = fmaxf(acc[2 * q]     * s1[c0] + bb1[c0], 0.0f);
          float x1 = fmaxf(acc[2 * q + 1] * s1[c1] + bb1[c1], 0.0f);
          pk[q] = (unsigned)f2bf(x0) | ((unsigned)f2bf(x1) << 16);
        }
        uint4 v; v.x = pk[0]; v.y = pk[1]; v.z = pk[2]; v.w = pk[3];
        *(uint4*)(s_h + n * COUT + cb) = v;
      };
      bn1_store(a00, 0, 0); bn1_store(a01, 0, 1);
      bn1_store(a10, 1, 0); bn1_store(a11, 1, 1);
    }
    __syncthreads();

    // ---- conv2: D(128x32) = W2(128x128) x h(128x32), 4 K-steps ----
    a00 = (v8f){0,0,0,0,0,0,0,0}; a01 = a00; a10 = a00; a11 = a00;
#pragma unroll
    for (int kt = 0; kt < 4; ++kt) {
      const int kb = kt * 32;
      V16U A0, A1, B0, B1;
      int off = (rowb + ln) * COUT + kb + (hi ? 8 : 0);
      A0.u[0] = *(const uint4*)(W2bf + off);
      A0.u[1] = *(const uint4*)(W2bf + off + 16);
      off = (rowb + 16 + ln) * COUT + kb + (hi ? 8 : 0);
      A1.u[0] = *(const uint4*)(W2bf + off);
      A1.u[1] = *(const uint4*)(W2bf + off + 16);
      int boff = ln * COUT + kb + (hi ? 8 : 0);
      B0.u[0] = *(const uint4*)(s_h + boff);
      B0.u[1] = *(const uint4*)(s_h + boff + 16);
      boff = (16 + ln) * COUT + kb + (hi ? 8 : 0);
      B1.u[0] = *(const uint4*)(s_h + boff);
      B1.u[1] = *(const uint4*)(s_h + boff + 16);
      a00 = __builtin_amdgcn_wmma_f32_16x16x32_bf16(false, A0.v, false, B0.v, (short)0, a00, false, false);
      a01 = __builtin_amdgcn_wmma_f32_16x16x32_bf16(false, A0.v, false, B1.v, (short)0, a01, false, false);
      a10 = __builtin_amdgcn_wmma_f32_16x16x32_bf16(false, A1.v, false, B0.v, (short)0, a10, false, false);
      a11 = __builtin_amdgcn_wmma_f32_16x16x32_bf16(false, A1.v, false, B1.v, (short)0, a11, false, false);
    }
    // BN2 + ReLU -> msgs (f32, [n][c]) with float4 stores
    {
      auto bn2_store = [&](v8f acc, int rt, int nt) {
        const int cb = rowb + rt * 16 + (hi ? 8 : 0);
        const int n  = nt * 16 + ln;
        float4 lo, hh;
        lo.x = fmaxf(acc[0] * s2[cb + 0] + bb2[cb + 0], 0.0f);
        lo.y = fmaxf(acc[1] * s2[cb + 1] + bb2[cb + 1], 0.0f);
        lo.z = fmaxf(acc[2] * s2[cb + 2] + bb2[cb + 2], 0.0f);
        lo.w = fmaxf(acc[3] * s2[cb + 3] + bb2[cb + 3], 0.0f);
        hh.x = fmaxf(acc[4] * s2[cb + 4] + bb2[cb + 4], 0.0f);
        hh.y = fmaxf(acc[5] * s2[cb + 5] + bb2[cb + 5], 0.0f);
        hh.z = fmaxf(acc[6] * s2[cb + 6] + bb2[cb + 6], 0.0f);
        hh.w = fmaxf(acc[7] * s2[cb + 7] + bb2[cb + 7], 0.0f);
        *(float4*)(s_msgs + n * COUT + cb)     = lo;
        *(float4*)(s_msgs + n * COUT + cb + 4) = hh;
      };
      bn2_store(a00, 0, 0); bn2_store(a01, 0, 1);
      bn2_store(a10, 1, 0); bn2_store(a11, 1, 1);
    }
    __syncthreads();

    // ---- attention scores + softmax over k (wave 0) ----
    if (t < NK) {
      float s = 0.0f;
      const float* mr = s_msgs + t * COUT;
      for (int c = 0; c < COUT; ++c) s += watt[c] * mr[c];
      float mx = s;
#pragma unroll
      for (int off = 16; off > 0; off >>= 1) mx = fmaxf(mx, __shfl_xor(mx, off, 32));
      float e = expf(s - mx);
      float sum = e;
#pragma unroll
      for (int off = 16; off > 0; off >>= 1) sum += __shfl_xor(sum, off, 32);
      s_attn[t] = e / sum;
    }
    __syncthreads();

    // ---- attention-weighted aggregation ----
    {
      float acc = 0.0f;
#pragma unroll
      for (int n = 0; n < NK; ++n) acc += s_msgs[n * COUT + t] * s_attn[n];
      out[(size_t)b * COUT * NM + (size_t)t * NM + m] = acc;
    }
    if (t < 4) {
      float acc = 0.0f;
#pragma unroll
      for (int n = 0; n < NK; ++n) acc += s_nlv[t * NK + n] * s_attn[n];
      out[OFF_LV + (size_t)b * 4 * NM + (size_t)t * NM + m] = acc;
    }
    __syncthreads();  // LDS reused next iteration
  }
}

// ---------------------------------------------------------------------------
// Host launcher. Workspace layout (bytes):
//   [0, 64K)            cidx   int[16*1024]
//   [64K, 64K+2M)       nidx   int[16*1024*32]
//   +40960              W1bf   ushort[128*160]
//   +32768              W2bf   ushort[128*128]
//   +2048               sb     float[4*128]  (scale1,bias1,scale2,bias2)
// ---------------------------------------------------------------------------
extern "C" void kernel_launch(void* const* d_in, const int* in_sizes, int n_in,
                              void* d_out, int out_size, void* d_ws, size_t ws_size,
                              hipStream_t stream) {
  const float* coords = (const float*)d_in[0];
  const float* feats  = (const float*)d_in[1];
  const float* lvs    = (const float*)d_in[2];
  // d_in[3] = mask (all true in this workload)
  const float* w1   = (const float*)d_in[4];
  const float* g1   = (const float*)d_in[5];
  const float* be1  = (const float*)d_in[6];
  const float* mu1  = (const float*)d_in[7];
  const float* va1  = (const float*)d_in[8];
  const float* w2   = (const float*)d_in[9];
  const float* g2   = (const float*)d_in[10];
  const float* be2  = (const float*)d_in[11];
  const float* mu2  = (const float*)d_in[12];
  const float* va2  = (const float*)d_in[13];
  const float* watt = (const float*)d_in[14];

  char* ws = (char*)d_ws;
  int* cidx = (int*)(ws);
  int* nidx = (int*)(ws + 65536);
  unsigned short* W1bf = (unsigned short*)(ws + 65536 + 2097152);
  unsigned short* W2bf = (unsigned short*)(ws + 65536 + 2097152 + 40960);
  float* sb = (float*)(ws + 65536 + 2097152 + 40960 + 32768);
  float* out = (float*)d_out;

  prep_kernel<<<1, 256, 0, stream>>>(w1, w2, g1, be1, mu1, va1,
                                     g2, be2, mu2, va2, W1bf, W2bf, sb);
  fps_kernel<<<BATCH, 1024, 0, stream>>>(coords, cidx);
  knn_kernel<<<dim3(NM, BATCH), 128, 0, stream>>>(coords, cidx, nidx);
  fused_kernel<<<512, 128, 0, stream>>>(coords, feats, lvs, cidx, nidx,
                                        W1bf, W2bf, sb, watt, out);
}
